// BURvNN_8091718386193
// MI455X (gfx1250) — matrix operand order repaired and checked by reading, not compiled, not run on hardware
//
#include <hip/hip_runtime.h>
#include <hip/hip_bf16.h>
#include <math.h>

#define N_NODES  16384
#define INP_DIM  5000
#define HID      64
#define N_CLASS  4
#define MAX_LVL  10
#define N_LEVELS (MAX_LVL + 1)
#define ROWTILES (N_NODES / 16)     // 1024
#define KSPLIT   4                  // K-chunks for the per-level adj GEMM
#define BSTRIDE  68                 // u32 row stride of packed B panel (16B aligned, bank-spread)
#define PANELSZ  (16 * BSTRIDE)     // u32s per B panel

typedef __attribute__((ext_vector_type(16))) __bf16   v16bf;
typedef __attribute__((ext_vector_type(8)))  float    v8f;
typedef __attribute__((ext_vector_type(8)))  unsigned v8u;

#define WMMA_BF16(a, b, c) \
  __builtin_amdgcn_wmma_f32_16x16x32_bf16(false, (a), false, (b), (short)0, (c), false, false)

// ---------------------------------------------------------------------------
// CDNA5 16-bit A layout (16x32, MxK): lane L holds row M = L%16; lanes 0-15
// hold K = {0..7, 16..23}, lanes 16-31 hold K = {8..15, 24..31}, 2/VGPR.
// B layout (32x16, KxN): lane L holds col N = L%16; lanes 0-15 hold K={0..7,16..23}
// (2/VGPR), lanes 16-31 hold K={8..15,24..31}.
// ---------------------------------------------------------------------------
static __device__ __forceinline__ int kofA(int v, int half) {
  return (v < 4 ? 2 * v : 16 + 2 * (v - 4)) + 8 * half;
}

static __device__ __forceinline__ unsigned pack_bf16(float f0, float f1) {
  union { unsigned u; __bf16 h[2]; } w;
  w.h[0] = (__bf16)f0;
  w.h[1] = (__bf16)f1;
  return w.u;
}

// Unguarded full-tile A fragment: two contiguous 8-float runs (32B aligned).
static __device__ __forceinline__ v16bf a_frag_full(const float* __restrict__ rowp,
                                                    int k0, int half) {
  const float4* lo = (const float4*)(rowp + k0 + 8 * half);
  const float4* hi = (const float4*)(rowp + k0 + 16 + 8 * half);
  float4 l0 = lo[0], l1 = lo[1], h0 = hi[0], h1 = hi[1];
  v16bf a;
  a[0] = (__bf16)l0.x;  a[1] = (__bf16)l0.y;  a[2]  = (__bf16)l0.z;  a[3]  = (__bf16)l0.w;
  a[4] = (__bf16)l1.x;  a[5] = (__bf16)l1.y;  a[6]  = (__bf16)l1.z;  a[7]  = (__bf16)l1.w;
  a[8] = (__bf16)h0.x;  a[9] = (__bf16)h0.y;  a[10] = (__bf16)h0.z;  a[11] = (__bf16)h0.w;
  a[12] = (__bf16)h1.x; a[13] = (__bf16)h1.y; a[14] = (__bf16)h1.z; a[15] = (__bf16)h1.w;
  return a;
}

// Guarded tail A fragment (runs once per kernel at most).
static __device__ __forceinline__ v16bf a_frag_tail(const float* __restrict__ rowp,
                                                    int k0, int kHi, int half) {
  v16bf a;
#pragma unroll
  for (int v = 0; v < 8; ++v) {
    int k = k0 + kofA(v, half);
    float f0 = (k     < kHi) ? rowp[k]     : 0.0f;
    float f1 = (k + 1 < kHi) ? rowp[k + 1] : 0.0f;
    a[2 * v]     = (__bf16)f0;
    a[2 * v + 1] = (__bf16)f1;
  }
  return a;
}

static __device__ __forceinline__ v16bf a_frag_lds(const float* __restrict__ s, // [16][64]
                                                   int m, int k0, int half) {
  v16bf a;
#pragma unroll
  for (int v = 0; v < 8; ++v) {
    int k = k0 + kofA(v, half);
    a[2 * v]     = (__bf16)s[m * HID + k];
    a[2 * v + 1] = (__bf16)s[m * HID + k + 1];
  }
  return a;
}

// B fragment from the pre-packed bf16 panel: pair p = K pair (2p, 2p+1).
static __device__ __forceinline__ v16bf b_frag_packed(const unsigned* __restrict__ s,
                                                      int col, int half) {
  v8u t;
#pragma unroll
  for (int v = 0; v < 8; ++v)
    t[v] = s[((v < 4 ? v : v + 4) + 4 * half) * BSTRIDE + col];
  return __builtin_bit_cast(v16bf, t);
}

static __device__ __forceinline__ v16bf b_frag_global(const float* __restrict__ B, // [K][64]
                                                      int k0, int col, int half) {
  v16bf b;
#pragma unroll
  for (int v = 0; v < 8; ++v) {
    int k = k0 + kofA(v, half);
    b[2 * v]     = (__bf16)B[k * HID + col];
    b[2 * v + 1] = (__bf16)B[(k + 1) * HID + col];
  }
  return b;
}

// Per-thread B-panel staging: thread covers K pair (2sp, 2sp+1), cols scb..scb+7.
static __device__ __forceinline__ void panel_load_full(const float* __restrict__ B,
                                                       int kg, int scb,
                                                       float* f0, float* f1) {
  const float4* r0 = (const float4*)(B + (long long)kg * HID + scb);
  const float4* r1 = (const float4*)(B + (long long)(kg + 1) * HID + scb);
  float4 a0 = r0[0], a1 = r0[1], b0 = r1[0], b1 = r1[1];
  f0[0] = a0.x; f0[1] = a0.y; f0[2] = a0.z; f0[3] = a0.w;
  f0[4] = a1.x; f0[5] = a1.y; f0[6] = a1.z; f0[7] = a1.w;
  f1[0] = b0.x; f1[1] = b0.y; f1[2] = b0.z; f1[3] = b0.w;
  f1[4] = b1.x; f1[5] = b1.y; f1[6] = b1.z; f1[7] = b1.w;
}

static __device__ __forceinline__ void panel_load_tail(const float* __restrict__ B,
                                                       int kg, int kHi, int scb,
                                                       float* f0, float* f1) {
#pragma unroll
  for (int u = 0; u < 8; ++u) {
    f0[u] = (kg     < kHi) ? B[(long long)kg * HID + scb + u]       : 0.0f;
    f1[u] = (kg + 1 < kHi) ? B[(long long)(kg + 1) * HID + scb + u] : 0.0f;
  }
}

static __device__ __forceinline__ void panel_store(unsigned* __restrict__ p,
                                                   int sp, int scb,
                                                   const float* f0, const float* f1) {
#pragma unroll
  for (int u = 0; u < 8; ++u)
    p[sp * BSTRIDE + scb + u] = pack_bf16(f0[u], f1[u]);
}

static __device__ __forceinline__ float sigmoidf(float x) {
  return 1.0f / (1.0f + __expf(-x));
}

// ---------------------------------------------------------------------------
// C[M,64] (chunk slab) = gather(A)[M, kLo:kHi] @ B[kLo:kHi, 64], bf16 WMMA,
// fp32 accumulate. 128 threads = 4 waves; wave w owns output cols [16w,16w+16).
// blockIdx.y selects a K-chunk writing its own C slab (fixed reduction order).
// Steady-state loop is branch-free: full tiles only, invalid rows read row 0
// (results discarded at the guarded store), B panel double-buffered in LDS as
// packed bf16 pairs with a single barrier per iteration. Tail tile handled
// once after the loop.
// ---------------------------------------------------------------------------
__global__ __launch_bounds__(128)
void gemm_n64_kernel(const float* __restrict__ A, long long lda,
                     const float* __restrict__ B,
                     float* __restrict__ C, long long cChunkStride,
                     const int* __restrict__ rows,     // optional A-row gather
                     int Mstatic, const int* __restrict__ Mdyn,
                     int K, int kChunk) {
  __shared__ unsigned sBp[2 * PANELSZ];
  const int M = Mdyn ? *Mdyn : Mstatic;
  const int tile = blockIdx.x;
  if (tile * 16 >= M) return;

  const int kLo = blockIdx.y * kChunk;
  const int kHi = (kLo + kChunk < K) ? (kLo + kChunk) : K;
  const int kFull = kLo + ((kHi - kLo) & ~31);   // end of full 32-wide tiles
  C += (long long)blockIdx.y * cChunkStride;

  const int tid  = threadIdx.x;
  const int lane = tid & 31;
  const int wave = tid >> 5;
  const int half = lane >> 4;
  const int m    = lane & 15;
  const int col  = wave * 16 + m;

  const int gr = tile * 16 + m;
  // Invalid rows alias row 0 (safe memory); their results are never stored.
  const int ar = (gr < M) ? (rows ? rows[gr] : gr) : 0;
  const float* rowp = A + (long long)ar * lda;

  // B-panel staging coords: K pair row sp, cols scb..scb+7.
  const int sp  = tid >> 3;        // 0..15
  const int scb = (tid & 7) * 8;   // 0,8,...,56

  v8f c = {};

  if (kFull > kLo) {
    // stage first panel
    {
      float f0[8], f1[8];
      panel_load_full(B, kLo + 2 * sp, scb, f0, f1);
      panel_store(sBp, sp, scb, f0, f1);
    }
    __syncthreads();

    int cur = 0;
    for (int k0 = kLo; k0 < kFull; k0 += 32) {
      const bool hasNext = (k0 + 32 < kFull);
      float f0[8], f1[8];
      if (hasNext) panel_load_full(B, k0 + 32 + 2 * sp, scb, f0, f1);  // overlap
      if (k0 + 256 < kFull) __builtin_prefetch((const void*)(rowp + k0 + 256), 0, 0);

      v16bf a = a_frag_full(rowp, k0, half);
      v16bf b = b_frag_packed(sBp + cur * PANELSZ, col, half);
      c = WMMA_BF16(a, b, c);

      if (hasNext) panel_store(sBp + (cur ^ 1) * PANELSZ, sp, scb, f0, f1);
      __syncthreads();      // writes target the other panel; reads preceded this
      cur ^= 1;
    }

    if (kFull < kHi) {       // guarded tail tile (embed GEMM only)
      {
        float f0[8], f1[8];
        panel_load_tail(B, kFull + 2 * sp, kHi, scb, f0, f1);
        panel_store(sBp, sp, scb, f0, f1);
      }
      __syncthreads();
      v16bf a = a_frag_tail(rowp, kFull, kHi, half);
      v16bf b = b_frag_packed(sBp, col, half);
      c = WMMA_BF16(a, b, c);
    }
  } else if (kLo < kHi) {    // chunk smaller than one tile
    {
      float f0[8], f1[8];
      panel_load_tail(B, kLo + 2 * sp, kHi, scb, f0, f1);
      panel_store(sBp, sp, scb, f0, f1);
    }
    __syncthreads();
    v16bf a = a_frag_tail(rowp, kLo, kHi, half);
    v16bf b = b_frag_packed(sBp, col, half);
    c = WMMA_BF16(a, b, c);
  }

  // C/D layout: VGPR v, lanes 0-15 -> row v, lanes 16-31 -> row v+8; col = lane%16
#pragma unroll
  for (int v = 0; v < 8; ++v) {
    const int go = tile * 16 + v + 8 * half;
    if (go < M) C[(long long)go * HID + col] = c[v];
  }
}

// ---------------------------------------------------------------------------
// Fused GRU gate kernel for one level: for compacted rows t < M:
//   hs = sum of KSPLIT partial panels (fixed order -> deterministic)
//   r = sigmoid(xr + hs@Ur), z = sigmoid(xz + hs@Uz)
//   hhat = tanh(xh + (hs*r)@Uh),  h[row] = (1-z)*hs + z*hhat
// ---------------------------------------------------------------------------
__global__ __launch_bounds__(128)
void gru_gate_kernel(const float* __restrict__ hs_part, long long chunkStride,
                     const int* __restrict__ rows,
                     const int* __restrict__ Mdyn,
                     const float* __restrict__ xr,
                     const float* __restrict__ xz,
                     const float* __restrict__ xh,
                     const float* __restrict__ Ur,
                     const float* __restrict__ Uz,
                     const float* __restrict__ Uh,
                     float* __restrict__ h) {
  __shared__ float sHS[16 * HID];
  __shared__ float sQ[16 * HID];
  const int M = *Mdyn;
  const int tile = blockIdx.x;
  if (tile * 16 >= M) return;

  const int tid = threadIdx.x;
  {
    const int r  = tid >> 3;
    const int cb = (tid & 7) * 8;
    const int go = tile * 16 + r;
    float acc[8];
#pragma unroll
    for (int u = 0; u < 8; ++u) acc[u] = 0.0f;
    if (go < M) {
#pragma unroll
      for (int cch = 0; cch < KSPLIT; ++cch) {
        const float* src = hs_part + (long long)cch * chunkStride
                         + (long long)go * HID + cb;
#pragma unroll
        for (int u = 0; u < 8; ++u) acc[u] += src[u];
      }
    }
#pragma unroll
    for (int u = 0; u < 8; ++u) sHS[r * HID + cb + u] = acc[u];
  }
  __syncthreads();

  const int lane = tid & 31;
  const int wave = tid >> 5;
  const int half = lane >> 4;
  const int m    = lane & 15;
  const int col  = wave * 16 + m;

  v8f cr = {}, cz = {};
#pragma unroll
  for (int k0 = 0; k0 < HID; k0 += 32) {
    v16bf a = a_frag_lds(sHS, m, k0, half);
    cr = WMMA_BF16(a, b_frag_global(Ur, k0, col, half), cr);
    cz = WMMA_BF16(a, b_frag_global(Uz, k0, col, half), cz);
  }

  float rj[8], zj[8], hsv[8];
  int arow[8];
  bool ok[8];
#pragma unroll
  for (int v = 0; v < 8; ++v) {
    const int rl = v + 8 * half;
    const int go = tile * 16 + rl;
    ok[v]   = go < M;
    arow[v] = ok[v] ? rows[go] : 0;
    const float xrv = ok[v] ? xr[(long long)arow[v] * HID + col] : 0.0f;
    const float xzv = ok[v] ? xz[(long long)arow[v] * HID + col] : 0.0f;
    hsv[v] = sHS[rl * HID + col];
    rj[v] = sigmoidf(xrv + cr[v]);
    zj[v] = sigmoidf(xzv + cz[v]);
    sQ[rl * HID + col] = hsv[v] * rj[v];
  }
  __syncthreads();

  v8f ch = {};
#pragma unroll
  for (int k0 = 0; k0 < HID; k0 += 32) {
    v16bf a = a_frag_lds(sQ, m, k0, half);
    ch = WMMA_BF16(a, b_frag_global(Uh, k0, col, half), ch);
  }

#pragma unroll
  for (int v = 0; v < 8; ++v) {
    if (ok[v]) {
      const float xhv  = xh[(long long)arow[v] * HID + col];
      const float hhat = tanhf(xhv + ch[v]);
      h[(long long)arow[v] * HID + col] = (1.0f - zj[v]) * hsv[v] + zj[v] * hhat;
    }
  }
}

// ---------------------------------------------------------------------------
__global__ void init_kernel(float* __restrict__ h, int nh, int* __restrict__ counts) {
  const int i = blockIdx.x * blockDim.x + threadIdx.x;
  if (i < nh) h[i] = 0.0f;
  if (i < N_LEVELS) counts[i] = 0;
}

__global__ void build_levels_kernel(const int* __restrict__ level,
                                    int* __restrict__ counts,
                                    int* __restrict__ rows) {
  const int i = blockIdx.x * blockDim.x + threadIdx.x;
  if (i < N_NODES) {
    const int j = level[i];
    const int p = atomicAdd(&counts[j], 1);
    rows[j * N_NODES + p] = i;
  }
}

// Deterministic two-stage root reduction: fixed row ranges, fixed add order.
__global__ __launch_bounds__(64)
void root_partial_kernel(const float* __restrict__ h, const int* __restrict__ level,
                         float* __restrict__ partial) {
  const int b = blockIdx.x;     // 128 blocks
  const int t = threadIdx.x;    // 64 threads = hidden col
  const int span = N_NODES / 128;
  float acc = 0.0f;
  for (int r = b * span; r < b * span + span; ++r)
    if (level[r] == 0) acc += h[(long long)r * HID + t];
  partial[b * HID + t] = acc;
}

__global__ __launch_bounds__(64)
void logits_kernel(const float* __restrict__ partial, const float* __restrict__ dec_w,
                   const float* __restrict__ dec_b, float* __restrict__ out) {
  __shared__ float root[HID];
  const int t = threadIdx.x;
  float acc = 0.0f;
  for (int b = 0; b < 128; ++b) acc += partial[b * HID + t];
  root[t] = acc;
  __syncthreads();
  if (t < N_CLASS) {
    float l = dec_b[t];
    for (int k = 0; k < HID; ++k) l += root[k] * dec_w[k * N_CLASS + t];
    out[t] = l;
  }
}

// ---------------------------------------------------------------------------
extern "C" void kernel_launch(void* const* d_in, const int* in_sizes, int n_in,
                              void* d_out, int out_size, void* d_ws, size_t ws_size,
                              hipStream_t stream) {
  (void)in_sizes; (void)n_in; (void)out_size; (void)ws_size;
  const float* adj   = (const float*)d_in[0];
  const float* tfidf = (const float*)d_in[1];
  const int*   level = (const int*)  d_in[2];
  const float* E_w   = (const float*)d_in[3];
  const float* Wr    = (const float*)d_in[4];
  const float* Wz    = (const float*)d_in[5];
  const float* Ur    = (const float*)d_in[6];
  const float* Uz    = (const float*)d_in[7];
  const float* Wh    = (const float*)d_in[8];
  const float* Uh    = (const float*)d_in[9];
  const float* dec_w = (const float*)d_in[10];
  const float* dec_b = (const float*)d_in[11];
  float* out = (float*)d_out;

  // workspace carve-up (256B aligned)
  char* ws = (char*)d_ws;
  size_t off = 0;
  auto alloc = [&](size_t bytes) {
    void* p = ws + off;
    off += (bytes + 255) & ~(size_t)255;
    return p;
  };
  const size_t mat = (size_t)N_NODES * HID * sizeof(float);   // 4 MB
  float* h       = (float*)alloc(mat);
  float* x_hat   = (float*)alloc(mat);
  float* xr      = (float*)alloc(mat);
  float* xz      = (float*)alloc(mat);
  float* xh      = (float*)alloc(mat);
  float* hs_part = (float*)alloc(mat * KSPLIT);                // per-chunk partial hs
  float* partial = (float*)alloc((size_t)128 * HID * sizeof(float));
  int*   counts  = (int*)alloc(N_LEVELS * sizeof(int));
  int*   rowsbuf = (int*)alloc((size_t)N_LEVELS * N_NODES * sizeof(int));

  const long long CHUNK_STRIDE = (long long)N_NODES * HID;

  // 1) zero h + per-level counts (ws is poisoned, not re-poisoned between replays)
  init_kernel<<<(N_NODES * HID + 255) / 256, 256, 0, stream>>>(h, N_NODES * HID, counts);

  // 2) compact node ids by level (each adj row then touched exactly once overall)
  build_levels_kernel<<<(N_NODES + 255) / 256, 256, 0, stream>>>(level, counts, rowsbuf);

  // 3) x_hat = tfidf @ E_w   (K = 5000: 156 full tiles + one guarded tail)
  gemm_n64_kernel<<<dim3(ROWTILES, 1), 128, 0, stream>>>(
      tfidf, (long long)INP_DIM, E_w, x_hat, 0,
      nullptr, N_NODES, nullptr, INP_DIM, INP_DIM);
  // 4) loop-invariant gate projections
  gemm_n64_kernel<<<dim3(ROWTILES, 1), 128, 0, stream>>>(
      x_hat, (long long)HID, Wr, xr, 0, nullptr, N_NODES, nullptr, HID, HID);
  gemm_n64_kernel<<<dim3(ROWTILES, 1), 128, 0, stream>>>(
      x_hat, (long long)HID, Wz, xz, 0, nullptr, N_NODES, nullptr, HID, HID);
  gemm_n64_kernel<<<dim3(ROWTILES, 1), 128, 0, stream>>>(
      x_hat, (long long)HID, Wh, xh, 0, nullptr, N_NODES, nullptr, HID, HID);

  // 5) bottom-up levels: K-split gathered adj GEMM + fused GRU gates.
  //    KSPLIT chunks x ~count/16 tiles keeps enough WGPs streaming adj to
  //    approach the 23.3 TB/s roofline even though each level is small.
  for (int j = MAX_LVL; j >= 0; --j) {
    gemm_n64_kernel<<<dim3(ROWTILES, KSPLIT), 128, 0, stream>>>(
        adj, (long long)N_NODES, h, hs_part, CHUNK_STRIDE,
        rowsbuf + j * N_NODES, 0, counts + j, N_NODES, N_NODES / KSPLIT);
    gru_gate_kernel<<<ROWTILES, 128, 0, stream>>>(
        hs_part, CHUNK_STRIDE, rowsbuf + j * N_NODES, counts + j,
        xr, xz, xh, Ur, Uz, Uh, h);
  }

  // 6) root reduction + logits
  root_partial_kernel<<<128, 64, 0, stream>>>(h, level, partial);
  logits_kernel<<<1, 64, 0, stream>>>(partial, dec_w, dec_b, out);
}
